// RGCNLayer_85521388798377
// MI455X (gfx1250) — matrix operand order repaired and checked by reading, not compiled, not run on hardware
//
#include <hip/hip_runtime.h>

typedef __attribute__((ext_vector_type(16))) _Float16 v16h;
typedef __attribute__((ext_vector_type(8)))  float    v8f;

#define N_NODES 50000
#define N_EDGES 600000
#define D 128
#define N_REL 8
#define WG_PER_REL 128

// workspace byte offsets (all 16B aligned)
#define OFF_XF16   0UL            // 50000*128 f16            = 12,800,000 B
#define OFF_BFRAG  12800000UL     // 9 slots * 128*128 f16    =    294,912 B
#define OFF_DEG    13094912UL     // 8*50000 u32              =  1,600,000 B
#define OFF_SORT   14694912UL     // 600000 u32               =  2,400,000 B
#define OFF_CTRL   17094912UL     // 32 u32 control block
// ctrl[0..7]=rel counts, ctrl[8..16]=rel offsets (excl scan), ctrl[17..24]=cursors

__device__ inline void atomAddF32(float* p, float v) { unsafeAtomicAdd(p, v); }

__global__ void k_zero(unsigned* deg, unsigned* ctrl) {
    int i = blockIdx.x * 256 + threadIdx.x;
    if (i < N_REL * N_NODES) deg[i] = 0u;
    if (i < 32) ctrl[i] = 0u;
}

__global__ void k_cvt_x(const float* __restrict__ x, _Float16* __restrict__ xf) {
    int i = blockIdx.x * 256 + threadIdx.x;       // one float4 per thread
    if (i >= (N_NODES * D) / 4) return;
    float4 v = ((const float4*)x)[i];
    _Float16 h[4] = { (_Float16)v.x, (_Float16)v.y, (_Float16)v.z, (_Float16)v.w };
    *(uint2*)(xf + (size_t)i * 4) = *(const uint2*)h;
}

// Repack weights into WMMA B-fragment order.
// B layout (16x16x32 f16): lane L holds K = kb*32 + (L/16)*16 + e, N = nb*16 + (L%16)
// frag = s*32 + nb*4 + kb ; element index = frag*512 + lane*16 + e
__global__ void k_repack(const float* __restrict__ W, const float* __restrict__ sW,
                         _Float16* __restrict__ bf) {
    int gid = blockIdx.x * 256 + threadIdx.x;
    if (gid >= 288 * 512) return;
    int e    = gid & 15;
    int lane = (gid >> 4) & 31;
    int frag = gid >> 9;
    int kb = frag & 3;
    int nb = (frag >> 2) & 7;
    int s  = frag >> 5;                  // 0..7 relations, 8 = self weight
    int K = kb * 32 + (lane >> 4) * 16 + e;
    int N = nb * 16 + (lane & 15);
    float v = (s < 8) ? W[((size_t)s * D + K) * D + N] : sW[(size_t)K * D + N];
    bf[gid] = (_Float16)v;
}

__global__ void k_count(const long long* __restrict__ ei, const long long* __restrict__ et,
                        unsigned* __restrict__ deg, unsigned* __restrict__ ctrl) {
    __shared__ unsigned lc[N_REL];
    int tid = threadIdx.x;
    if (tid < N_REL) lc[tid] = 0u;
    __syncthreads();
    int e = blockIdx.x * 256 + tid;
    if (e < N_EDGES) {
        int r = (int)et[e];
        int d = (int)ei[N_EDGES + e];
        atomicAdd(&deg[(size_t)r * N_NODES + d], 1u);
        atomicAdd(&lc[r], 1u);
    }
    __syncthreads();
    if (tid < N_REL) atomicAdd(&ctrl[tid], lc[tid]);
}

__global__ void k_scan(unsigned* ctrl) {
    unsigned acc = 0;
    for (int r = 0; r < N_REL; ++r) {
        ctrl[8 + r]  = acc;   // offsets
        ctrl[17 + r] = acc;   // cursors
        acc += ctrl[r];
    }
    ctrl[16] = acc;
}

__global__ void k_scatter(const long long* __restrict__ et, unsigned* __restrict__ ctrl,
                          unsigned* __restrict__ sorted) {
    int e = blockIdx.x * 256 + threadIdx.x;
    if (e >= N_EDGES) return;
    int r = (int)et[e];
    unsigned pos = atomicAdd(&ctrl[17 + r], 1u);
    sorted[pos] = (unsigned)e;
}

// out = x @ self_weight + bias  (initializes out before edge accumulation)
__global__ __launch_bounds__(128) void k_self(const _Float16* __restrict__ xf,
                                              const _Float16* __restrict__ bf,
                                              const float* __restrict__ bias,
                                              float* __restrict__ out) {
    int tile = blockIdx.x;                 // 3125 tiles of 16 nodes
    int wave = threadIdx.x >> 5;
    int lane = threadIdx.x & 31;
    int half = lane >> 4;
    int n    = lane & 15;
    int m    = lane & 15;

    #pragma unroll
    for (int i = 0; i < 2; ++i) {
        int nb = wave * 2 + i;
        v8f c = {};
        #pragma unroll
        for (int kb = 0; kb < 4; ++kb) {
            const _Float16* ap = xf + ((size_t)(tile * 16 + m)) * D + kb * 32 + half * 16;
            v16h a;
            ((uint4*)&a)[0] = *(const uint4*)ap;
            ((uint4*)&a)[1] = *(const uint4*)(ap + 8);
            const _Float16* bp = bf + ((size_t)(256 + nb * 4 + kb) * 512 + lane * 16);
            v16h b;
            ((uint4*)&b)[0] = *(const uint4*)bp;
            ((uint4*)&b)[1] = *(const uint4*)(bp + 8);
            c = __builtin_amdgcn_wmma_f32_16x16x32_f16(false, a, false, b,
                                                       (short)0, c, false, false);
        }
        float bv = bias[nb * 16 + n];
        #pragma unroll
        for (int j = 0; j < 8; ++j) {
            int node = tile * 16 + j + half * 8;
            out[(size_t)node * D + nb * 16 + n] = c[j] + bv;
        }
    }
}

// Per-relation edge-tile GEMM: 16 edges x 128 out via WMMA, then scaled atomic scatter.
__global__ __launch_bounds__(128) void k_edge(const long long* __restrict__ ei,
                                              const _Float16* __restrict__ xf,
                                              const _Float16* __restrict__ bf,
                                              const unsigned* __restrict__ deg,
                                              const unsigned* __restrict__ ctrl,
                                              const unsigned* __restrict__ sorted,
                                              float* __restrict__ out) {
    __shared__ __align__(16) _Float16 Ash[16 * 136];   // padded stride: bank-conflict-free
    __shared__ int   srcSh[16];
    __shared__ int   dstSh[16];
    __shared__ float normSh[16];

    int r    = blockIdx.x / WG_PER_REL;
    int wg   = blockIdx.x % WG_PER_REL;
    int tid  = threadIdx.x;
    int wave = tid >> 5;
    int lane = tid & 31;
    int half = lane >> 4;
    int n    = lane & 15;
    int m    = lane & 15;

    unsigned cnt = ctrl[r];
    unsigned off = ctrl[8 + r];
    unsigned nT  = (cnt + 15u) >> 4;

    // B fragments for this relation held in registers across all tiles
    v16h Breg[2][4];
    #pragma unroll
    for (int i = 0; i < 2; ++i) {
        #pragma unroll
        for (int kb = 0; kb < 4; ++kb) {
            int nb = wave * 2 + i;
            const _Float16* bp = bf + ((size_t)((r * 8 + nb) * 4 + kb) * 512 + lane * 16);
            ((uint4*)&Breg[i][kb])[0] = *(const uint4*)bp;
            ((uint4*)&Breg[i][kb])[1] = *(const uint4*)(bp + 8);
        }
    }

    for (unsigned t = wg; t < nT; t += WG_PER_REL) {
        if (tid < 16) {
            unsigned slot = t * 16 + tid;
            int s = 0, d = 0; float nm = 0.0f;
            if (slot < cnt) {
                unsigned e = sorted[off + slot];
                s = (int)ei[e];
                d = (int)ei[N_EDGES + e];
                unsigned dg = deg[(size_t)r * N_NODES + d];
                nm = 1.0f / (float)(dg ? dg : 1u);
            }
            srcSh[tid] = s; dstSh[tid] = d; normSh[tid] = nm;
        }
        __syncthreads();

        // gather 16 source rows (16 x 128 f16) into LDS: 256 uint4, 2 per thread
        #pragma unroll
        for (int i = 0; i < 2; ++i) {
            int slot = tid + i * 128;
            int row  = slot >> 4;
            int part = slot & 15;
            uint4 v = ((const uint4*)(xf + (size_t)srcSh[row] * D))[part];
            *(uint4*)(&Ash[row * 136 + part * 8]) = v;
        }
        __syncthreads();

        #pragma unroll
        for (int i = 0; i < 2; ++i) {
            int nb = wave * 2 + i;
            v8f c = {};
            #pragma unroll
            for (int kb = 0; kb < 4; ++kb) {
                const _Float16* ap = &Ash[m * 136 + kb * 32 + half * 16];
                v16h a;
                ((uint4*)&a)[0] = *(const uint4*)ap;
                ((uint4*)&a)[1] = *(const uint4*)(ap + 8);
                c = __builtin_amdgcn_wmma_f32_16x16x32_f16(false, a, false, Breg[i][kb],
                                                           (short)0, c, false, false);
            }
            #pragma unroll
            for (int j = 0; j < 8; ++j) {
                int M = j + half * 8;
                float v = c[j] * normSh[M];
                atomAddF32(out + (size_t)dstSh[M] * D + nb * 16 + n, v);
            }
        }
        __syncthreads();
    }
}

extern "C" void kernel_launch(void* const* d_in, const int* in_sizes, int n_in,
                              void* d_out, int out_size, void* d_ws, size_t ws_size,
                              hipStream_t stream) {
    const float*     x    = (const float*)d_in[0];
    const long long* ei   = (const long long*)d_in[1];   // [2, E] int64
    const long long* et   = (const long long*)d_in[2];   // [E]    int64
    const float*     W    = (const float*)d_in[3];       // [8,128,128]
    const float*     sW   = (const float*)d_in[4];       // [128,128]
    const float*     bias = (const float*)d_in[5];       // [128]
    float*           out  = (float*)d_out;

    char* ws = (char*)d_ws;
    _Float16* xf     = (_Float16*)(ws + OFF_XF16);
    _Float16* bfrag  = (_Float16*)(ws + OFF_BFRAG);
    unsigned* deg    = (unsigned*)(ws + OFF_DEG);
    unsigned* sorted = (unsigned*)(ws + OFF_SORT);
    unsigned* ctrl   = (unsigned*)(ws + OFF_CTRL);

    (void)in_sizes; (void)n_in; (void)out_size; (void)ws_size;

    k_zero   <<<(N_REL * N_NODES + 255) / 256, 256, 0, stream>>>(deg, ctrl);
    k_cvt_x  <<<((N_NODES * D / 4) + 255) / 256, 256, 0, stream>>>(x, xf);
    k_repack <<<(288 * 512) / 256, 256, 0, stream>>>(W, sW, bfrag);
    k_count  <<<(N_EDGES + 255) / 256, 256, 0, stream>>>(ei, et, deg, ctrl);
    k_scan   <<<1, 1, 0, stream>>>(ctrl);
    k_scatter<<<(N_EDGES + 255) / 256, 256, 0, stream>>>(et, ctrl, sorted);
    k_self   <<<N_NODES / 16, 128, 0, stream>>>(xf, bfrag, bias, out);
    k_edge   <<<N_REL * WG_PER_REL, 128, 0, stream>>>(ei, xf, bfrag, deg, ctrl, sorted, out);
}
